// SelfAttention_12008728560273
// MI455X (gfx1250) — matrix operand order repaired and checked by reading, not compiled, not run on hardware
//
#include <hip/hip_runtime.h>
#include <math.h>

// Problem constants (match reference)
#define BB 2
#define LL 2048
#define DD 512
#define HH 8
#define HDIM 64
#define NLAYER 4
#define KDIM 512          // contraction dim of every GEMM (D)
#define THREE_D 1536

typedef float v2f  __attribute__((ext_vector_type(2)));
typedef float v8f  __attribute__((ext_vector_type(8)));
typedef __bf16 v16bf __attribute__((ext_vector_type(16)));

// ---------------------------------------------------------------------------
// WMMA wrappers
//  f32 : D(16x16) = A(16x4) * B(4x16) + C         (exact fp32, attention)
//  bf16: D(16x16) = A(16x32) * B(32x16) + C       (split-3 fp32 emulation, GEMMs)
// ---------------------------------------------------------------------------
__device__ __forceinline__ v8f wmma_f32(v2f a, v2f b, v8f c) {
  return __builtin_amdgcn_wmma_f32_16x16x4_f32(false, a, false, b, (short)0, c,
                                               false, false);
}
__device__ __forceinline__ v8f wmma_bf16(v16bf a, v16bf b, v8f c) {
  return __builtin_amdgcn_wmma_f32_16x16x32_bf16(false, a, false, b, (short)0, c,
                                                 false, false);
}

// ---------------------------------------------------------------------------
// CDNA5 async global->LDS copy (ASYNCcnt path), inline asm
// ---------------------------------------------------------------------------
__device__ __forceinline__ void async_b128(unsigned lds_addr, const void* gaddr) {
  asm volatile("global_load_async_to_lds_b128 %0, %1, off"
               :: "v"(lds_addr), "v"(gaddr) : "memory");
}
__device__ __forceinline__ void wait_async0() {
  asm volatile("s_wait_asynccnt 0x0" ::: "memory");
}
__device__ __forceinline__ unsigned lds_addr_of(const void* p) {
  return (unsigned)(unsigned long long)p;   // addrspace(3) ptr -> byte offset
}

// bf16 round-to-nearest-even helpers (integer tricks; no __bf16 scalar math)
__device__ __forceinline__ unsigned short bf16_rne(float v) {
  unsigned b = __float_as_uint(v);
  return (unsigned short)((b + 0x7FFFu + ((b >> 16) & 1u)) >> 16);
}
__device__ __forceinline__ float bf16_to_f32(unsigned short h) {
  return __uint_as_float(((unsigned)h) << 16);
}

// ---------------------------------------------------------------------------
// x = base + timing_signal   (one block per (b,l) row)
// ---------------------------------------------------------------------------
__global__ __launch_bounds__(256) void timing_kernel(const float* __restrict__ base,
                                                     float* __restrict__ x) {
  const int bl = blockIdx.x;
  const int l  = bl & (LL - 1);
  const float pos = (float)l - 1.0f;               // seq_start = 1
  const float inc = logf(10000.0f) / 255.0f;       // nt-1 = 255
  const size_t off = (size_t)bl * DD;
#pragma unroll
  for (int e = 0; e < 2; ++e) {
    const int d = threadIdx.x + e * 256;
    const int j = (d < 256) ? d : d - 256;
    const float inv = expf(-(float)j * inc);
    const float st  = pos * inv;
    const float ts  = (d < 256) ? sinf(st) : cosf(st);
    x[off + d] = base[off + d] + ts;
  }
}

// ---------------------------------------------------------------------------
// fp32 -> (bf16 hi, bf16 lo) split, elementwise
// ---------------------------------------------------------------------------
__global__ __launch_bounds__(256) void split_kernel(const float* __restrict__ src,
                                                    unsigned short* __restrict__ hi,
                                                    unsigned short* __restrict__ lo,
                                                    int n) {
  for (int i = blockIdx.x * blockDim.x + threadIdx.x; i < n;
       i += gridDim.x * blockDim.x) {
    const float v = src[i];
    const unsigned short h = bf16_rne(v);
    hi[i] = h;
    lo[i] = bf16_rne(v - bf16_to_f32(h));
  }
}

// ---------------------------------------------------------------------------
// Split-3 bf16 GEMM: C[M,N] = A[M,512]*W[N,512]^T + bias (+ epilogue)
//   EPI 0: bias | EPI 1: bias+res | EPI 2: relu(bias'd)+res
// block 256 (8 waves, 2x4), tile 128x64, K-slab 32, async double-buffered LDS.
// bf16 tile row stride = 40 shorts (80B): 16B-aligned chunks, conflict-free.
// ---------------------------------------------------------------------------
#define RS 40

union BFrag { unsigned d[8]; v16bf v; };

__device__ __forceinline__ v16bf load_frag(const unsigned short* row, int o0, int o1) {
  BFrag f;
  const uint4 a = *(const uint4*)(row + o0);
  const uint4 b = *(const uint4*)(row + o1);
  f.d[0] = a.x; f.d[1] = a.y; f.d[2] = a.z; f.d[3] = a.w;
  f.d[4] = b.x; f.d[5] = b.y; f.d[6] = b.z; f.d[7] = b.w;
  return f.v;
}

template <int EPI>
__global__ __launch_bounds__(256) void gemm_bf16_kernel(
    const unsigned short* __restrict__ Ah, const unsigned short* __restrict__ Al,
    const unsigned short* __restrict__ Wh, const unsigned short* __restrict__ Wl,
    const float* __restrict__ bias, const float* __restrict__ res,
    float* __restrict__ Cout, int N) {
  __shared__ __align__(16) unsigned short sAh[2][128 * RS];
  __shared__ __align__(16) unsigned short sAl[2][128 * RS];
  __shared__ __align__(16) unsigned short sWh[2][64 * RS];
  __shared__ __align__(16) unsigned short sWl[2][64 * RS];

  const int tid  = threadIdx.x;
  const int lane = tid & 31;
  const int wid  = tid >> 5;
  const int wm   = wid >> 2;          // 0..1 : 64 rows
  const int wn   = wid & 3;           // 0..3 : 16 cols
  const int half = lane >> 4;
  const int r    = lane & 15;

  const int m0 = blockIdx.y * 128;
  const int n0 = blockIdx.x * 64;

  // stage one K-slab (32 wide) of all four tiles with async copies
  auto stage = [&](int buf, int kb) {
#pragma unroll
    for (int p = 0; p < 2; ++p) {                 // A tiles: 512 16B chunks each
      const int c   = p * 256 + tid;
      const int row = c >> 2;
      const int seg = (c & 3) * 8;                // shorts
      const size_t g = (size_t)(m0 + row) * KDIM + kb + seg;
      const unsigned loff = (unsigned)(row * RS + seg) * 2u;
      async_b128(lds_addr_of(&sAh[buf][0]) + loff, Ah + g);
      async_b128(lds_addr_of(&sAl[buf][0]) + loff, Al + g);
    }
    {                                             // W tiles: 256 chunks each
      const int row = tid >> 2;
      const int seg = (tid & 3) * 8;
      const size_t g = (size_t)(n0 + row) * KDIM + kb + seg;
      const unsigned loff = (unsigned)(row * RS + seg) * 2u;
      async_b128(lds_addr_of(&sWh[buf][0]) + loff, Wh + g);
      async_b128(lds_addr_of(&sWl[buf][0]) + loff, Wl + g);
    }
  };

  v8f acc[4] = {};
  stage(0, 0);
  int buf = 0;

  for (int kb = 0; kb < KDIM; kb += 32) {
    wait_async0();
    __syncthreads();
    if (kb + 32 < KDIM) stage(buf ^ 1, kb + 32);

    // B fragments (this wave's 16 output columns)
    const unsigned short* wrow_h = &sWh[buf][(wn * 16 + r) * RS];
    const unsigned short* wrow_l = &sWl[buf][(wn * 16 + r) * RS];
    const v16bf bhi = load_frag(wrow_h, half * 8, 16 + half * 8);
    const v16bf blo = load_frag(wrow_l, half * 8, 16 + half * 8);

#pragma unroll
    for (int mt = 0; mt < 4; ++mt) {
      const unsigned short* arow_h = &sAh[buf][(wm * 64 + mt * 16 + r) * RS];
      const unsigned short* arow_l = &sAl[buf][(wm * 64 + mt * 16 + r) * RS];
      const v16bf ahi = load_frag(arow_h, half * 8, 16 + half * 8);
      const v16bf alo = load_frag(arow_l, half * 8, 16 + half * 8);
      acc[mt] = wmma_bf16(ahi, bhi, acc[mt]);   // hi*hi
      acc[mt] = wmma_bf16(ahi, blo, acc[mt]);   // hi*lo
      acc[mt] = wmma_bf16(alo, bhi, acc[mt]);   // lo*hi
    }
    buf ^= 1;
  }

  // epilogue
  const int col = n0 + wn * 16 + r;
  const float bv = bias[col];
#pragma unroll
  for (int mt = 0; mt < 4; ++mt) {
    const int mbase = m0 + wm * 64 + mt * 16 + half * 8;
#pragma unroll
    for (int v = 0; v < 8; ++v) {
      const int row = mbase + v;
      float val = acc[mt][v] + bv;
      if (EPI == 1) val += res[(size_t)row * N + col];
      if (EPI == 2) val = fmaxf(val, 0.0f) + res[(size_t)row * N + col];
      Cout[(size_t)row * N + col] = val;
    }
  }
}

// ---------------------------------------------------------------------------
// Flash attention, exact fp32 WMMA. grid=(L/64,H,B), block=128 (4 waves).
// K/V 16x64 tiles async double-buffered in LDS (stride 68: 16B-aligned rows).
// ---------------------------------------------------------------------------
#define LDT 18
#define KVS 68

__global__ __launch_bounds__(128) void attn_kernel(const float* __restrict__ qkv,
                                                   float* __restrict__ obuf) {
  __shared__ __align__(16) float Kt[2][16 * KVS];
  __shared__ __align__(16) float Vt[2][16 * KVS];
  __shared__ float Sw[4][16 * LDT];
  __shared__ float Aux[4][16];

  const int tid  = threadIdx.x;
  const int lane = tid & 31;
  const int wid  = tid >> 5;
  const int half = lane >> 4;
  const int r    = lane & 15;

  const int h = blockIdx.y;
  const int b = blockIdx.z;
  const int qrow0 = blockIdx.x * 64 + wid * 16;

  auto stage_kv = [&](int buf, int krow0) {
#pragma unroll
    for (int p = 0; p < 2; ++p) {                  // 256 16B chunks per matrix
      const int c   = p * 128 + tid;
      const int row = c >> 4;
      const int seg = (c & 15) * 4;                // floats
      const size_t g = (size_t)(b * LL + krow0 + row) * THREE_D + h * HDIM + seg;
      const unsigned loff = (unsigned)(row * KVS + seg) * 4u;
      async_b128(lds_addr_of(&Kt[buf][0]) + loff, qkv + g + DD);
      async_b128(lds_addr_of(&Vt[buf][0]) + loff, qkv + g + 2 * DD);
    }
  };

  // preload Q fragments (K=64 -> 16 k-steps), pre-scaled by 1/sqrt(HD)
  v2f qf[16];
  {
    const float* qrow = qkv + (size_t)(b * LL + qrow0 + r) * THREE_D + h * HDIM;
#pragma unroll
    for (int ks = 0; ks < 16; ++ks) {
      v2f q = *(const v2f*)(qrow + ks * 4 + half * 2);
      q.x *= 0.125f; q.y *= 0.125f;
      qf[ks] = q;
    }
  }

  v8f o[4] = {};
  float m_run = -3.0e38f;
  float l_run = 0.0f;
  float* Sp = &Sw[wid][0];

  stage_kv(0, 0);
  int cur = 0;

  for (int kt = 0; kt < LL / 16; ++kt) {
    wait_async0();
    __syncthreads();
    if (kt + 1 < LL / 16) stage_kv(cur ^ 1, (kt + 1) * 16);
    const float* Kc = &Kt[cur][0];
    const float* Vc = &Vt[cur][0];

    // S = (Q/8) * K^T : 16 fp32 WMMAs over K=64
    v8f s = {};
#pragma unroll
    for (int ks = 0; ks < 16; ++ks) {
      const v2f bk = *(const v2f*)&Kc[r * KVS + ks * 4 + half * 2];
      s = wmma_f32(qf[ks], bk, s);
    }
#pragma unroll
    for (int v = 0; v < 8; ++v)
      Sp[(v + 8 * half) * LDT + r] = s[v];
    __builtin_amdgcn_wave_barrier();

    // online softmax: lane m (0..15) owns row m
    if (lane < 16) {
      float rowmax = -3.0e38f;
#pragma unroll
      for (int j = 0; j < 16; ++j) rowmax = fmaxf(rowmax, Sp[lane * LDT + j]);
      const float m_new = fmaxf(m_run, rowmax);
      const float corr  = __expf(m_run - m_new);
      float ssum = 0.0f;
#pragma unroll
      for (int j = 0; j < 16; ++j) {
        const float p = __expf(Sp[lane * LDT + j] - m_new);
        Sp[lane * LDT + j] = p;
        ssum += p;
      }
      l_run = l_run * corr + ssum;
      m_run = m_new;
      Aux[wid][lane] = corr;
    }
    __builtin_amdgcn_wave_barrier();

    float av[8];
#pragma unroll
    for (int v = 0; v < 8; ++v) av[v] = Aux[wid][v + 8 * half];
#pragma unroll
    for (int nt = 0; nt < 4; ++nt)
#pragma unroll
      for (int v = 0; v < 8; ++v) o[nt][v] *= av[v];

    // O += P * V : 16 fp32 WMMAs
#pragma unroll
    for (int ks = 0; ks < 4; ++ks) {
      const int koff = ks * 4 + half * 2;
      const v2f pa = *(const v2f*)&Sp[r * LDT + koff];
#pragma unroll
      for (int nt = 0; nt < 4; ++nt) {
        v2f bv;
        bv.x = Vc[(koff + 0) * KVS + nt * 16 + r];
        bv.y = Vc[(koff + 1) * KVS + nt * 16 + r];
        o[nt] = wmma_f32(pa, bv, o[nt]);
      }
    }
    cur ^= 1;
  }

  if (lane < 16) Aux[wid][lane] = 1.0f / l_run;
  __builtin_amdgcn_wave_barrier();
  float lv[8];
#pragma unroll
  for (int v = 0; v < 8; ++v) lv[v] = Aux[wid][v + 8 * half];
#pragma unroll
  for (int nt = 0; nt < 4; ++nt)
#pragma unroll
    for (int v = 0; v < 8; ++v) {
      const int m = v + 8 * half;
      obuf[(size_t)(b * LL + qrow0 + m) * DD + h * HDIM + nt * 16 + r] =
          o[nt][v] * lv[v];
    }
}

// ---------------------------------------------------------------------------
// Batch-global reductions + scalar-affine normalize
// ---------------------------------------------------------------------------
__global__ void zero4_kernel(float* p) {
  if (threadIdx.x < 2 * BB) p[threadIdx.x] = 0.0f;
}

__global__ __launch_bounds__(256) void reduce_kernel(const float* __restrict__ y,
                                                     float* __restrict__ sums) {
  __shared__ float red[512];
  const int n = LL * DD;
  const int bidx = blockIdx.y;
  const float* p = y + (size_t)bidx * n;
  float s = 0.0f, s2 = 0.0f;
  for (int i = blockIdx.x * blockDim.x + threadIdx.x; i < n;
       i += gridDim.x * blockDim.x) {
    const float v = p[i];
    s += v; s2 += v * v;
  }
  red[threadIdx.x] = s;
  red[256 + threadIdx.x] = s2;
  __syncthreads();
  for (int off = 128; off > 0; off >>= 1) {
    if (threadIdx.x < off) {
      red[threadIdx.x] += red[threadIdx.x + off];
      red[256 + threadIdx.x] += red[256 + threadIdx.x + off];
    }
    __syncthreads();
  }
  if (threadIdx.x == 0) {
    atomicAdd(&sums[bidx * 2 + 0], red[0]);
    atomicAdd(&sums[bidx * 2 + 1], red[256]);
  }
}

__global__ __launch_bounds__(256) void norm_kernel(const float* __restrict__ y,
                                                   float* __restrict__ out,
                                                   const float* __restrict__ sums,
                                                   const float* __restrict__ scale_arr,
                                                   const float* __restrict__ bias_arr,
                                                   int layer) {
  const int n = LL * DD;
  const int bidx = blockIdx.y;
  const float invn = 1.0f / (float)n;
  const float m   = sums[bidx * 2 + 0] * invn;
  const float var = sums[bidx * 2 + 1] * invn - m * m;
  const float inv_sd = rsqrtf(fmaxf(var, 1.0e-30f));
  const float sc = scale_arr[layer];
  const float bi = bias_arr[layer];
  const size_t base = (size_t)bidx * n;
  for (int i = blockIdx.x * blockDim.x + threadIdx.x; i < n;
       i += gridDim.x * blockDim.x)
    out[base + i] = (y[base + i] - m) * inv_sd * sc + bi;
}

// ---------------------------------------------------------------------------
// Launch
// ---------------------------------------------------------------------------
extern "C" void kernel_launch(void* const* d_in, const int* in_sizes, int n_in,
                              void* d_out, int out_size, void* d_ws, size_t ws_size,
                              hipStream_t stream) {
  const float* base     = (const float*)d_in[0];
  const float* qkv_w    = (const float*)d_in[1];
  const float* qkv_b    = (const float*)d_in[2];
  const float* f1_w     = (const float*)d_in[3];
  const float* f1_b     = (const float*)d_in[4];
  const float* f2_w     = (const float*)d_in[5];
  const float* f2_b     = (const float*)d_in[6];
  const float* n1_scale = (const float*)d_in[7];
  const float* n1_bias  = (const float*)d_in[8];
  const float* n2_scale = (const float*)d_in[9];
  const float* n2_bias  = (const float*)d_in[10];
  float* out = (float*)d_out;

  // ---- workspace layout ----
  char* wsb = (char*)d_ws;
  size_t off = 0;
  auto alloc_f = [&](size_t n) { float* p = (float*)(wsb + off);
                                 off += n * sizeof(float); return p; };
  float* xbuf   = alloc_f((size_t)BB * LL * DD);
  float* qkvbuf = alloc_f((size_t)BB * LL * THREE_D);
  float* obuf   = alloc_f((size_t)BB * LL * DD);
  float* ybuf   = alloc_f((size_t)BB * LL * DD);
  float* hbuf   = alloc_f((size_t)BB * LL * DD);
  float* sums   = alloc_f(8);
  off = (off + 255) & ~(size_t)255;
  auto alloc_h = [&](size_t n) { unsigned short* p = (unsigned short*)(wsb + off);
                                 off += n * sizeof(unsigned short); return p; };
  const size_t NQKVW = (size_t)NLAYER * THREE_D * KDIM;
  const size_t NFFW  = (size_t)NLAYER * DD * KDIM;
  const size_t NACT  = (size_t)BB * LL * DD;
  unsigned short* qkvw_h = alloc_h(NQKVW);
  unsigned short* qkvw_l = alloc_h(NQKVW);
  unsigned short* f1w_h  = alloc_h(NFFW);
  unsigned short* f1w_l  = alloc_h(NFFW);
  unsigned short* f2w_h  = alloc_h(NFFW);
  unsigned short* f2w_l  = alloc_h(NFFW);
  unsigned short* act_h  = alloc_h(NACT);
  unsigned short* act_l  = alloc_h(NACT);

  const int M = BB * LL;   // 4096

  // split all weights to bf16 hi/lo once
  split_kernel<<<3072, 256, 0, stream>>>(qkv_w, qkvw_h, qkvw_l, (int)NQKVW);
  split_kernel<<<1024, 256, 0, stream>>>(f1_w, f1w_h, f1w_l, (int)NFFW);
  split_kernel<<<1024, 256, 0, stream>>>(f2_w, f2w_h, f2w_l, (int)NFFW);

  timing_kernel<<<BB * LL, 256, 0, stream>>>(base, xbuf);

  for (int i = 0; i < NLAYER; ++i) {
    const size_t wq = (size_t)i * THREE_D * KDIM;
    const size_t wf = (size_t)i * DD * KDIM;

    // qkv = x @ qkv_w^T + qkv_b
    split_kernel<<<2048, 256, 0, stream>>>(xbuf, act_h, act_l, (int)NACT);
    gemm_bf16_kernel<0><<<dim3(THREE_D / 64, M / 128), 256, 0, stream>>>(
        act_h, act_l, qkvw_h + wq, qkvw_l + wq,
        qkv_b + (size_t)i * THREE_D, nullptr, qkvbuf, THREE_D);

    // o = softmax(QK^T/8) V   (exact fp32 WMMA)
    attn_kernel<<<dim3(LL / 64, HH, BB), 128, 0, stream>>>(qkvbuf, obuf);

    // y = o @ f1_w^T + f1_b + x
    split_kernel<<<2048, 256, 0, stream>>>(obuf, act_h, act_l, (int)NACT);
    gemm_bf16_kernel<1><<<dim3(DD / 64, M / 128), 256, 0, stream>>>(
        act_h, act_l, f1w_h + wf, f1w_l + wf,
        f1_b + (size_t)i * DD, xbuf, ybuf, DD);

    // h = norm(y; n1)
    zero4_kernel<<<1, 32, 0, stream>>>(sums);
    reduce_kernel<<<dim3(128, BB), 256, 0, stream>>>(ybuf, sums);
    norm_kernel<<<dim3(1024, BB), 256, 0, stream>>>(ybuf, hbuf, sums,
                                                    n1_scale, n1_bias, i);

    // y = relu(h @ f2_w^T + f2_b) + h
    split_kernel<<<2048, 256, 0, stream>>>(hbuf, act_h, act_l, (int)NACT);
    gemm_bf16_kernel<2><<<dim3(DD / 64, M / 128), 256, 0, stream>>>(
        act_h, act_l, f2w_h + wf, f2w_l + wf,
        f2_b + (size_t)i * DD, hbuf, ybuf, DD);

    // x = norm(y; n2)  (last layer -> d_out)
    zero4_kernel<<<1, 32, 0, stream>>>(sums);
    reduce_kernel<<<dim3(128, BB), 256, 0, stream>>>(ybuf, sums);
    float* dst = (i == NLAYER - 1) ? out : xbuf;
    norm_kernel<<<dim3(1024, BB), 256, 0, stream>>>(ybuf, dst, sums,
                                                    n2_scale, n2_bias, i);
  }
}
